// CellularNN_50268297232533
// MI455X (gfx1250) — matrix operand order repaired and checked by reading, not compiled, not run on hardware
//
#include <hip/hip_runtime.h>

typedef __attribute__((ext_vector_type(16))) _Float16 v16h;
typedef __attribute__((ext_vector_type(8)))  _Float16 v8h;
typedef __attribute__((ext_vector_type(8)))  float    v8f;

#define K1PAD     160   // 144 padded up to a multiple of 32
#define SP_STRIDE 168   // sP row stride (halfs) -> conflict-free b128 reads
#define SH_STRIDE 520   // sH row stride (halfs)
#define SDX_STRIDE 68   // sDx row stride (floats)

// ---------------------------------------------------------------------------
// Weight prep: fc1_w (144x512 f32) -> w1t[n=512][k=160] f16 (zero-padded K),
//              fc2_w (512x64  f32) -> w2t[c=64][k=512] f16.
// Transposed so a WMMA B-fragment lane (column N, 16 contiguous K) is one
// contiguous 32-byte read.
// ---------------------------------------------------------------------------
__global__ __launch_bounds__(256) void nca_prep(const float* __restrict__ fc1_w,
                                                const float* __restrict__ fc2_w,
                                                _Float16* __restrict__ w1t,
                                                _Float16* __restrict__ w2t) {
  const int i = blockIdx.x * 256 + threadIdx.x;
  if (i < 512 * K1PAD) {
    const int n = i / K1PAD, k = i % K1PAD;
    w1t[i] = (k < 144) ? (_Float16)fc1_w[k * 512 + n] : (_Float16)0.0f;
  } else {
    const int j = i - 512 * K1PAD;
    if (j < 64 * 512) {
      const int c = j >> 9, k = j & 511;
      w2t[j] = (_Float16)fc2_w[k * 64 + c];
    }
  }
}

// Build a 16-bit A-fragment: lane holds row M = lane&15, halfs
// K = b..b+15 with b = 16*(lane>>4). Two 16B loads + shuffle.
__device__ __forceinline__ v16h mkfrag(const _Float16* p) {
  v8h lo = *(const v8h*)p;
  v8h hi = *(const v8h*)(p + 8);
  return __builtin_shufflevector(lo, hi, 0,1,2,3,4,5,6,7,8,9,10,11,12,13,14,15);
}

__device__ __forceinline__ v8f wmma16(v16h a, v16h b, v8f c) {
  // (neg_a, A, neg_b, B, c_mod, C, reuse_a, reuse_b)
  return __builtin_amdgcn_wmma_f32_16x16x32_f16(false, a, false, b, (short)0, c,
                                                false, false);
}

// ---------------------------------------------------------------------------
// Fused NCA step: perceive -> fc1+ReLU -> fc2 -> residual -> LayerNorm.
// One WG = 256 threads (8 waves) handles 32 consecutive pixels of one image.
// ---------------------------------------------------------------------------
__global__ __launch_bounds__(256) void nca_main(
    const float* __restrict__ x,
    const float* __restrict__ fc1_b, const float* __restrict__ fc2_b,
    const float* __restrict__ ln_w,  const float* __restrict__ ln_b,
    const _Float16* __restrict__ w1t, const _Float16* __restrict__ w2t,
    float* __restrict__ out)
{
  __shared__ alignas(32) _Float16 sP[32 * SP_STRIDE];  // perception, f16
  __shared__ alignas(32) _Float16 sH[32 * SH_STRIDE];  // hidden, f16
  __shared__ float sDx[32 * SDX_STRIDE];               // dx, f32

  const int t    = threadIdx.x;
  const int wave = t >> 5;
  const int lane = t & 31;
  const int ln16 = lane & 15;
  const int half = lane >> 4;

  const int p0  = blockIdx.x * 32;   // global pixel base (tile stays in-row)
  const int b   = p0 >> 12;          // 4096 pixels per image
  const int hw0 = p0 & 4095;
  const int xb0 = b * 262144;        // b * 64ch * 4096px

  // ---- phase 0: build perception tile in LDS (32 px x 160 feat, f16) ----
  {
    const int p    = t >> 3;         // pixel within tile, 0..31
    const int part = t & 7;          // feature slice 20*part..+19
    const int hw = hw0 + p;
    const int h = hw >> 6, w = hw & 63;
    const int offr[8] = {-1,-1,-1, 0, 0, 1, 1, 1};
    const int offc[8] = {-1, 0, 1,-1, 1,-1, 0, 1};
    #pragma unroll
    for (int q = 0; q < 20; ++q) {
      const int c = part * 20 + q;
      float v = 0.0f;
      if (c < 64) {
        v = x[xb0 + c * 4096 + hw];
      } else if (c < 144) {
        const int j  = (c - 64) / 10;
        const int cc = (c - 64) % 10;
        const int hh = h + offr[j], ww = w + offc[j];
        if (hh >= 0 && hh < 64 && ww >= 0 && ww < 64)
          v = x[xb0 + cc * 4096 + hh * 64 + ww];
      }
      sP[p * SP_STRIDE + c] = (_Float16)v;
    }
  }
  __syncthreads();

  // ---- phase 1: GEMM1 [32 x 160] x [160 x 512], wave owns 64 N-cols ----
  v8f acc[2][4];
  #pragma unroll
  for (int mt = 0; mt < 2; ++mt)
    #pragma unroll
    for (int nt = 0; nt < 4; ++nt)
      acc[mt][nt] = {};

  const int n0 = wave * 64;
  #pragma unroll
  for (int k0 = 0; k0 < K1PAD; k0 += 32) {
    const v16h a0 = mkfrag(&sP[ln16        * SP_STRIDE + k0 + 16 * half]);
    const v16h a1 = mkfrag(&sP[(16 + ln16) * SP_STRIDE + k0 + 16 * half]);
    #pragma unroll
    for (int nt = 0; nt < 4; ++nt) {
      const v16h bf =
          *(const v16h*)&w1t[(n0 + nt * 16 + ln16) * K1PAD + k0 + 16 * half];
      acc[0][nt] = wmma16(a0, bf, acc[0][nt]);
      acc[1][nt] = wmma16(a1, bf, acc[1][nt]);
    }
  }

  // bias + ReLU -> f16 hidden in LDS. C layout: lane holds col N = ln16,
  // rows M = r + 8*half.
  #pragma unroll
  for (int nt = 0; nt < 4; ++nt) {
    const int n = n0 + nt * 16 + ln16;
    const float bias = fc1_b[n];
    #pragma unroll
    for (int r = 0; r < 8; ++r) {
      const int row = r + 8 * half;
      const float v0 = fmaxf(acc[0][nt][r] + bias, 0.0f);
      const float v1 = fmaxf(acc[1][nt][r] + bias, 0.0f);
      sH[row        * SH_STRIDE + n] = (_Float16)v0;
      sH[(16 + row) * SH_STRIDE + n] = (_Float16)v1;
    }
  }
  __syncthreads();

  // ---- phase 2: GEMM2 [32 x 512] x [512 x 64], one 16x16 tile per wave ----
  const int mt2 = wave >> 2;   // M tile (pixels)
  const int nt2 = wave & 3;    // N tile (out channels)
  v8f d = {};
  #pragma unroll
  for (int k0 = 0; k0 < 512; k0 += 32) {
    const v16h a = mkfrag(&sH[(mt2 * 16 + ln16) * SH_STRIDE + k0 + 16 * half]);
    const v16h bf =
        *(const v16h*)&w2t[(nt2 * 16 + ln16) * 512 + k0 + 16 * half];
    d = wmma16(a, bf, d);
  }
  {
    const int c = nt2 * 16 + ln16;
    const float bias = fc2_b[c];
    #pragma unroll
    for (int r = 0; r < 8; ++r)
      sDx[(mt2 * 16 + r + 8 * half) * SDX_STRIDE + c] = d[r] + bias;
  }
  __syncthreads();

  // ---- phase 3: residual + LayerNorm over 64 channels per pixel ----
  {
    const int p  = t >> 3;           // pixel 0..31 (8 threads per pixel)
    const int cb = (t & 7) * 8;      // this thread's 8 channels
    const int hw = hw0 + p;
    float v[8];
    float s = 0.0f;
    #pragma unroll
    for (int i = 0; i < 8; ++i) {
      const int c = cb + i;
      v[i] = x[xb0 + c * 4096 + hw] + sDx[p * SDX_STRIDE + c];
      s += v[i];
    }
    s += __shfl_xor(s, 1);
    s += __shfl_xor(s, 2);
    s += __shfl_xor(s, 4);
    const float mean = s * (1.0f / 64.0f);
    float q = 0.0f;
    #pragma unroll
    for (int i = 0; i < 8; ++i) { const float dv = v[i] - mean; q += dv * dv; }
    q += __shfl_xor(q, 1);
    q += __shfl_xor(q, 2);
    q += __shfl_xor(q, 4);
    const float inv = rsqrtf(q * (1.0f / 64.0f) + 1e-5f);
    #pragma unroll
    for (int i = 0; i < 8; ++i) {
      const int c = cb + i;
      out[xb0 + c * 4096 + hw] = (v[i] - mean) * inv * ln_w[c] + ln_b[c];
    }
  }
}

extern "C" void kernel_launch(void* const* d_in, const int* in_sizes, int n_in,
                              void* d_out, int out_size, void* d_ws, size_t ws_size,
                              hipStream_t stream) {
  (void)in_sizes; (void)n_in; (void)out_size; (void)ws_size;
  const float* x     = (const float*)d_in[0];
  const float* fc1_w = (const float*)d_in[1];
  const float* fc1_b = (const float*)d_in[2];
  const float* fc2_w = (const float*)d_in[3];
  const float* fc2_b = (const float*)d_in[4];
  const float* ln_w  = (const float*)d_in[5];
  const float* ln_b  = (const float*)d_in[6];
  float* out = (float*)d_out;

  _Float16* w1t = (_Float16*)d_ws;           // 512*160 halfs = 160 KB
  _Float16* w2t = w1t + 512 * K1PAD;         // 64*512 halfs  =  64 KB

  const int total = 512 * K1PAD + 64 * 512;
  nca_prep<<<(total + 255) / 256, 256, 0, stream>>>(fc1_w, fc2_w, w1t, w2t);

  // 131072 pixels / 32 per WG = 4096 workgroups, 256 threads (8 waves) each.
  nca_main<<<4096, 256, 0, stream>>>(x, fc1_b, fc2_b, ln_w, ln_b, w1t, w2t, out);
}
